// NLMP_54941221650459
// MI455X (gfx1250) — matrix operand order repaired and checked by reading, not compiled, not run on hardware
//
#include <hip/hip_runtime.h>

typedef _Float16 v16h __attribute__((ext_vector_type(16)));
typedef float    v8f  __attribute__((ext_vector_type(8)));
typedef float    v4f  __attribute__((ext_vector_type(4)));

#define BLOCK_THREADS 64
#define WAVES 2
#define EDGES_PER_WAVE 32

__global__ __launch_bounds__(BLOCK_THREADS)
void nlmp_fused(const float* __restrict__ x,
                const int*   __restrict__ edge_src,
                const int*   __restrict__ edge_dst,
                const float* __restrict__ edge_vec,
                const float* __restrict__ emb,
                const float* __restrict__ norm,
                const float* __restrict__ W1,
                const float* __restrict__ W2,
                float* __restrict__ out,
                int E)
{
    // ---- LDS (63.4 KB) ----
    __shared__ float W1s[160];                                   // 10x16
    __shared__ __align__(32) _Float16 W2h[14336];                // [j=0..895][k=0..15] f16
    __shared__ __align__(32) _Float16 zeropad[16];               // zero B rows (K=16..31)
    __shared__ __align__(16) float Ltab[WAVES][8][16][32];       // [wave][vec][u][edge]
    __shared__ float y1tab[WAVES][32][3];
    __shared__ float normtab[WAVES][32];
    __shared__ int   dsttab[WAVES][32];

    const int tid  = threadIdx.x;
    const int wave = tid >> 5;
    const int lane = tid & 31;

    // ---- stage W1 / W2 (transposed, f16) : coalesced global reads ----
    for (int i = tid; i < 160; i += BLOCK_THREADS) W1s[i] = W1[i];
    if (tid < 16) zeropad[tid] = (_Float16)0.0f;
    for (int k = 0; k < 16; ++k) {
        #pragma unroll 1
        for (int jj = 0; jj < 14; ++jj) {
            int j = jj * 64 + tid;
            W2h[j * 16 + k] = (_Float16)W2[k * 896 + j];
        }
    }
    __syncthreads();

    const int wave_base = (blockIdx.x * WAVES + wave) * EDGES_PER_WAVE;

    // ---- per-edge prep: one lane per edge (32 edges/wave) ----
    {
        const int eg_raw = wave_base + lane;
        const bool valid = (eg_raw < E);
        const int eg = valid ? eg_raw : 0;
        const int nsrc = edge_src[eg];
        const int ndst = edge_dst[eg];
        float vx = edge_vec[eg * 3 + 0];
        float vy = edge_vec[eg * 3 + 1];
        float vz = edge_vec[eg * 3 + 2];
        float rin = rsqrtf(vx * vx + vy * vy + vz * vz);
        float nx = vx * rin, ny = vy * rin, nz = vz * rin;
        const float S3 = 1.7320508075688772f;
        float y10 = S3 * nx, y11 = S3 * ny, y12 = S3 * nz;
        const float S15 = 3.872983346207417f;
        float y20 = S15 * nx * ny;
        float y21 = S15 * ny * nz;
        float y22 = 1.118033988749895f * (3.0f * nz * nz - 1.0f);
        float y23 = S15 * nx * nz;
        float y24 = 1.9364916731037085f * (nx * nx - ny * ny);
        const float C1  = 0.31622776601683794f;   // 1/(sqrt2*sqrt5)
        const float C2  = 0.18257418583505536f;   // 1/(sqrt6*sqrt5)
        const float C2b = 0.3651483716701107f;    // 2/(sqrt6*sqrt5)
        float T00 = -C2 * y22 + C1 * y24;
        float T01 =  C1 * y20;
        float T02 =  C1 * y23;
        float T10 =  C1 * y20;
        float T11 = -C2 * y22 - C1 * y24;
        float T12 =  C1 * y21;
        float T20 =  C1 * y23;
        float T21 =  C1 * y21;
        float T22 =  C2b * y22;
        const float I3 = 0.5773502691896258f;
        #pragma unroll
        for (int half = 0; half < 2; ++half) {
            const float* xp = x + (long)(half ? ndst : nsrc) * 32;
            #pragma unroll
            for (int up = 0; up < 8; ++up) {
                int u = half * 8 + up;
                float sv = xp[up];
                float a0 = xp[8 + up * 3 + 0];
                float a1 = xp[8 + up * 3 + 1];
                float a2 = xp[8 + up * 3 + 2];
                Ltab[wave][0][u][lane] = sv;                                    // xs
                Ltab[wave][1][u][lane] = I3 * (a0 * y10 + a1 * y11 + a2 * y12); // inv3*(xv.y1)
                Ltab[wave][2][u][lane] = I3 * a0;                               // inv3*xv_k
                Ltab[wave][3][u][lane] = I3 * a1;
                Ltab[wave][4][u][lane] = I3 * a2;
                Ltab[wave][5][u][lane] = a0 * T00 + a1 * T10 + a2 * T20;        // r_k
                Ltab[wave][6][u][lane] = a0 * T01 + a1 * T11 + a2 * T21;
                Ltab[wave][7][u][lane] = a0 * T02 + a1 * T12 + a2 * T22;
            }
        }
        y1tab[wave][lane][0] = y10;
        y1tab[wave][lane][1] = y11;
        y1tab[wave][lane][2] = y12;
        normtab[wave][lane] = valid ? norm[eg] : 0.0f;
        dsttab[wave][lane]  = ndst;
    }

    // ---- A fragments: h = 0.25*sqrt(2)*relu(emb@W1/sqrt(10)), f16, K upper half zero ----
    const int part  = lane >> 4;
    const int kbase = part * 8;
    auto make_a = [&](int elocal) -> v16h {
        int er = wave_base + elocal;
        int eh = er < E ? er : 0;
        const float ISQ10 = 0.31622776601683794f;
        const float HS = 0.25f * 1.4142135623730951f;
        float e0[10];
        #pragma unroll
        for (int r = 0; r < 10; ++r) e0[r] = emb[(long)eh * 10 + r];
        v16h a;
        #pragma unroll
        for (int k8 = 0; k8 < 16; ++k8) {
            float acc = 0.0f;
            #pragma unroll
            for (int r = 0; r < 10; ++r) acc += e0[r] * W1s[r * 16 + k8];
            acc *= ISQ10;
            acc = acc > 0.0f ? acc : 0.0f;
            a[k8] = (_Float16)(part == 0 ? HS * acc : 0.0f);
        }
        return a;
    };
    const v16h amat0 = make_a(lane & 15);
    const v16h amat1 = make_a(16 + (lane & 15));
    __syncthreads();

    // ---- main: 56 N-tiles, two 16-edge A fragments share each B tile ----
    const int n = lane & 15;
    const int wout  = n & 7;
    const int u_off = (n >> 3) & 1;
    const int mbase = part * 8;        // which 8 edges of a fragment this lane's C holds
    const int mq    = mbase >> 2;      // v4f index (0 or 2)

    const _Float16* bcol = (part == 0) ? &W2h[n * 16] : zeropad;
    const int bstep = (part == 0) ? 256 : 0;     // halves per 16-col tile

    float aS0[8], aS1[8], aG0[8], aG1[8], aC0[8], aC1[8];
    float aV00[8], aV01[8], aV10[8], aV11[8], aV20[8], aV21[8];
    #pragma unroll
    for (int v = 0; v < 8; ++v) {
        aS0[v]=0.f; aS1[v]=0.f; aG0[v]=0.f; aG1[v]=0.f; aC0[v]=0.f; aC1[v]=0.f;
        aV00[v]=0.f; aV01[v]=0.f; aV10[v]=0.f; aV11[v]=0.f; aV20[v]=0.f; aV21[v]=0.f;
    }

    auto fma8 = [](float (&acc)[8], const v8f& c, const v4f& la, const v4f& lb) {
        acc[0] += la[0]*c[0]; acc[1] += la[1]*c[1]; acc[2] += la[2]*c[2]; acc[3] += la[3]*c[3];
        acc[4] += lb[0]*c[4]; acc[5] += lb[1]*c[5]; acc[6] += lb[2]*c[6]; acc[7] += lb[3]*c[7];
    };

    auto pass1 = [&](int baseT, int vec, float (&a0)[8], float (&a1)[8]) {
        #pragma unroll 2
        for (int tt = 0; tt < 8; ++tt) {
            v16h bm = *(const v16h*)(bcol + (baseT + tt) * bstep);
            v8f c0 = {}, c1 = {};
            c0 = __builtin_amdgcn_wmma_f32_16x16x32_f16(false, amat0, false, bm,
                                                        (short)0, c0, false, false);
            c1 = __builtin_amdgcn_wmma_f32_16x16x32_f16(false, amat1, false, bm,
                                                        (short)0, c1, false, false);
            const int u = (tt << 1) | u_off;
            const v4f* Lp = (const v4f*)&Ltab[wave][vec][u][0];
            fma8(a0, c0, Lp[mq],     Lp[mq + 1]);
            fma8(a1, c1, Lp[mq + 4], Lp[mq + 5]);
        }
    };
    auto pass3 = [&](int baseT, int v0i, int v1i, int v2i) {
        #pragma unroll 1
        for (int tt = 0; tt < 8; ++tt) {
            v16h bm = *(const v16h*)(bcol + (baseT + tt) * bstep);
            v8f c0 = {}, c1 = {};
            c0 = __builtin_amdgcn_wmma_f32_16x16x32_f16(false, amat0, false, bm,
                                                        (short)0, c0, false, false);
            c1 = __builtin_amdgcn_wmma_f32_16x16x32_f16(false, amat1, false, bm,
                                                        (short)0, c1, false, false);
            const int u = (tt << 1) | u_off;
            const v4f* P0 = (const v4f*)&Ltab[wave][v0i][u][0];
            const v4f* P1 = (const v4f*)&Ltab[wave][v1i][u][0];
            const v4f* P2 = (const v4f*)&Ltab[wave][v2i][u][0];
            fma8(aV00, c0, P0[mq],     P0[mq + 1]);
            fma8(aV01, c1, P0[mq + 4], P0[mq + 5]);
            fma8(aV10, c0, P1[mq],     P1[mq + 1]);
            fma8(aV11, c1, P1[mq + 4], P1[mq + 5]);
            fma8(aV20, c0, P2[mq],     P2[mq + 1]);
            fma8(aV21, c1, P2[mq + 4], P2[mq + 5]);
        }
    };

    pass1( 0, 0, aS0, aS1);   // wa_s : xs
    pass1( 8, 1, aS0, aS1);   // wb_s : inv3*(xv.y1)
    pass1(16, 0, aG0, aG1);   // wa_g : xs
    pass1(24, 1, aG0, aG1);   // wb_g : inv3*(xv.y1)
    pass1(32, 0, aC0, aC1);   // wc   : xs
    pass3(40, 2, 3, 4);       // wd   : inv3*xv_k
    pass3(48, 5, 6, 7);       // we   : r_k

    // ---- fold even/odd-u lanes ----
    auto red = [](float (&a)[8]) {
        #pragma unroll
        for (int v = 0; v < 8; ++v) a[v] += __shfl_xor(a[v], 8, 32);
    };
    red(aS0); red(aS1); red(aG0); red(aG1); red(aC0); red(aC1);
    red(aV00); red(aV01); red(aV10); red(aV11); red(aV20); red(aV21);

    // ---- gate, scale, scatter-add ----
    if ((lane & 8) == 0) {
        const float G   = 1.593f;
        const float PWS = 0.17677669529663687f;
        const float PWV = 0.25f;
        const float I3  = 0.5773502691896258f;
        auto writeb = [&](const float (&S)[8], const float (&Gv)[8], const float (&Cc)[8],
                          const float (&V0)[8], const float (&V1)[8], const float (&V2)[8],
                          int ebase) {
            #pragma unroll
            for (int v = 0; v < 8; ++v) {
                int e = ebase + v;
                float nrm = normtab[wave][e];
                int dst   = dsttab[wave][e];
                float sact = G * tanhf(PWS * S[v]);
                float gate = G * tanhf(PWS * Gv[v]);
                float* op = out + (long)dst * 32;
                atomicAdd(op + wout, sact * nrm);
                float gn = gate * nrm * PWV;
                float vo0 = I3 * y1tab[wave][e][0] * Cc[v] + V0[v];
                float vo1 = I3 * y1tab[wave][e][1] * Cc[v] + V1[v];
                float vo2 = I3 * y1tab[wave][e][2] * Cc[v] + V2[v];
                atomicAdd(op + 8 + wout * 3 + 0, gn * vo0);
                atomicAdd(op + 8 + wout * 3 + 1, gn * vo1);
                atomicAdd(op + 8 + wout * 3 + 2, gn * vo2);
            }
        };
        writeb(aS0, aG0, aC0, aV00, aV10, aV20, mbase);
        writeb(aS1, aG1, aC1, aV01, aV11, aV21, 16 + mbase);
    }
}

extern "C" void kernel_launch(void* const* d_in, const int* in_sizes, int n_in,
                              void* d_out, int out_size, void* d_ws, size_t ws_size,
                              hipStream_t stream)
{
    const float* x        = (const float*)d_in[0];
    const int*   edge_src = (const int*)  d_in[1];
    const int*   edge_dst = (const int*)  d_in[2];
    const float* edge_vec = (const float*)d_in[3];
    const float* emb      = (const float*)d_in[4];
    const float* nrm      = (const float*)d_in[5];
    const float* W1       = (const float*)d_in[6];
    const float* W2       = (const float*)d_in[7];
    float* out = (float*)d_out;

    const int E = in_sizes[1];

    hipMemsetAsync(d_out, 0, (size_t)out_size * sizeof(float), stream);

    const int epb    = WAVES * EDGES_PER_WAVE;      // 64 edges per block
    const int blocks = (E + epb - 1) / epb;
    nlmp_fused<<<blocks, BLOCK_THREADS, 0, stream>>>(
        x, edge_src, edge_dst, edge_vec, emb, nrm, W1, W2, out, E);
}
